// LinearNavigator_1254130450669
// MI455X (gfx1250) — compile-verified, matrix operand order checked by reading, not run on hardware
//
#include <hip/hip_runtime.h>
#include <hip/hip_bf16.h>

// LSTM "LinearNavigator": B=1M rows, H=20, 2 LSTM layers + tiny projection.
// Roofline: 664 MB moved -> ~28.5us @ 23.3 TB/s HBM floor. ~10.5 GFLOP, 95% in
// shared-weight gate GEMMs [B,22]x[22,80] and [B,41]x[41,80] -> run those on
// v_wmma_f32_16x16x32_f16 (f16 in / f32 acc) so the kernel is HBM-bound.
// Rev 4: native v_tanh_f32 (gfx1250 TRANS op) for all sigmoid/tanh, guarded by
// __has_builtin with an exp-based fallback; halves the elementwise VALU work.
// Retains: pre-swizzled B-fragments (2x ds_load_b128), lane*10+it ownership
// (no div/mod, merged b64/b128 global traffic), unrolled hot loops.

typedef __attribute__((ext_vector_type(16))) _Float16 v16h;
typedef __attribute__((ext_vector_type(8)))  _Float16 v8h;
typedef __attribute__((ext_vector_type(8)))  float    v8f;

#define B_TOTAL 1048576
#define H 20
#define G 80          // 4*H gate width
#define K2 64         // layer-2 K padded: [h0(20),h_prev(20),1,zeros]
#define WAVES 4
#define RPW 16        // rows per wave (one 16xG gate tile)
#define RPB (WAVES*RPW)

#if __has_builtin(__builtin_amdgcn_tanhf)
__device__ __forceinline__ float ftanhf(float x)   { return __builtin_amdgcn_tanhf(x); }
__device__ __forceinline__ float fsigmoid(float x) {
    return __builtin_fmaf(0.5f, __builtin_amdgcn_tanhf(0.5f * x), 0.5f);
}
#else
__device__ __forceinline__ float fsigmoid(float x) { return 1.0f / (1.0f + __expf(-x)); }
__device__ __forceinline__ float ftanhf(float x)   { return 2.0f * fsigmoid(2.0f * x) - 1.0f; }
#endif

// 16-bit WMMA fragment K striping (ISA 7.12.2): half-wave hi, VGPR v holds the
// packed K pair starting at:
__device__ __forceinline__ int kpair(int v, int hi) {
    return ((v & 3) << 1) | ((v >> 2) << 4) | (hi << 3);
}

__device__ __forceinline__ v16h frag_from_lds(const _Float16* p /*16B aligned, 16 halves*/) {
    const v8h* pv = (const v8h*)p;
    v8h lo = pv[0], hi = pv[1];
    return __builtin_shufflevector(lo, hi, 0,1,2,3,4,5,6,7,8,9,10,11,12,13,14,15);
}

extern "C" __global__ void __launch_bounds__(128, 1)
lstm_navigator_kernel(const float* __restrict__ inp,
                      const float* __restrict__ hidden,   // [2,B,H]
                      const float* __restrict__ cell,     // [2,B,H]
                      const float* __restrict__ W_ih1,    // [80,2]
                      const float* __restrict__ W_hh1,    // [80,20]
                      const float* __restrict__ b_ih1,
                      const float* __restrict__ b_hh1,
                      const float* __restrict__ W_ih2,    // [80,20]
                      const float* __restrict__ W_hh2,    // [80,20]
                      const float* __restrict__ b_ih2,
                      const float* __restrict__ b_hh2,
                      const float* __restrict__ W_out,    // [1,20]
                      const float* __restrict__ b_out,    // [1]
                      float* __restrict__ dout)           // out(5B) | hid(40B) | cell(40B)
{
    // ---- LDS ----
    // Weight B-fragments in exact per-lane WMMA byte order:
    // BF[frag][lane][e] : e = 2v+odd -> (k = kpair(v,lane>>4)+odd, n = nt*16+(lane&15))
    __shared__ __align__(16) _Float16 BF1[5][32][16];        // layer1               (5 KB)
    __shared__ __align__(16) _Float16 BF2[10][32][16];       // layer2 [chunk*5+nt]  (10 KB)
    __shared__ float    WoutS[H];
    __shared__ float    boutS;
    __shared__ __align__(16) _Float16 stage[WAVES][RPW][K2]; // per-wave A staging   (8 KB)
    __shared__ float    gates[WAVES][RPW][G];                // per-wave gate tile   (20 KB)

    const int tid  = threadIdx.x;
    const int wv   = tid >> 5;
    const int lane = tid & 31;
    const int hi   = lane >> 4;     // half-wave
    const int lm   = lane & 15;     // M (A/D) or N (B) coordinate

    // ---- build swizzled weight fragments (block-cooperative, once, cold) ----
    for (int t = tid; t < 5 * 32 * 16; t += 128) {
        int nt = t >> 9, l = (t >> 4) & 31, e = t & 15;
        int k = kpair(e >> 1, l >> 4) + (e & 1);
        int n = nt * 16 + (l & 15);
        float w = 0.0f;
        if (k < 2)        w = W_ih1[n * 2 + k];
        else if (k < 22)  w = W_hh1[n * H + (k - 2)];
        else if (k == 22) w = b_ih1[n] + b_hh1[n];
        BF1[nt][l][e] = (_Float16)w;
    }
    for (int t = tid; t < 10 * 32 * 16; t += 128) {
        int f = t >> 9;                       // f = chunk*5 + nt
        int c = f / 5, nt = f % 5;
        int l = (t >> 4) & 31, e = t & 15;
        int k = 32 * c + kpair(e >> 1, l >> 4) + (e & 1);
        int n = nt * 16 + (l & 15);
        float w = 0.0f;
        if (k < 20)       w = W_ih2[n * H + k];
        else if (k < 40)  w = W_hh2[n * H + (k - 20)];
        else if (k == 40) w = b_ih2[n] + b_hh2[n];
        BF2[f][l][e] = (_Float16)w;
    }
    if (tid < H)  WoutS[tid] = W_out[tid];
    if (tid == 0) boutS = b_out[0];
    __syncthreads();

    const long rowBase = (long)blockIdx.x * RPB + (long)wv * RPW;
    float* outO = dout;                                  // [B,5]
    float* outH = dout + 5L  * B_TOTAL;                  // [2,B,H]
    float* outC = dout + 45L * B_TOTAL;                  // [2,B,H]

    // Per-lane element ownership: lane handles row em, j = ej0..ej0+9
    // (10 consecutive floats per lane -> merged wide loads/stores, no div/mod).
    const int  em   = lane >> 1;
    const int  ej0  = (lane & 1) * 10;
    const long base1 = rowBase * H + (long)lane * 10;    // == (rowBase+em)*H + ej0
    const long base2 = (long)B_TOTAL * H + base1;

    // ---- stage layer-1 K-vectors: [preproc(x)(2), h(20), 1, zeros] ----
    {   // preproc: 2 lanes per row
        int m = lane >> 1, c = lane & 1;
        float x  = inp[rowBase + m];
        float ax = fabsf(x);
        bool keep = ax >= 4.5399929762e-5f;              // exp(-10)
        float v;
        if (c == 0) v = keep ? (__logf(ax + 1e-8f) * 0.1f) : -1.0f;
        else        v = keep ? ((x > 0.0f) ? 1.0f : ((x < 0.0f) ? -1.0f : 0.0f))
                             : (22026.465794806718f * x);   // exp(10)*x
        stage[wv][m][c] = (_Float16)v;
    }
#pragma unroll
    for (int it = 0; it < 10; ++it)                      // h -> cols 2..21
        stage[wv][em][2 + ej0 + it] = (_Float16)hidden[base1 + it];
#pragma unroll
    for (int it = 0; it < 5; ++it) {                     // cols 22..31: bias-1 + pad
        int k = 22 + (lane & 1) * 5 + it;
        stage[wv][em][k] = (_Float16)((k == 22) ? 1.0f : 0.0f);
    }
    __syncthreads();

    // ---- layer 1: A fragment (2x b128) + 5 N-tiles of WMMA ----
    v16h a1;
    {
        const v8h* pa = (const v8h*)&stage[wv][lm][0];   // row = 64 halves, 16B chunks
        a1 = __builtin_shufflevector(pa[hi], pa[2 + hi],
                                     0,1,2,3,4,5,6,7,8,9,10,11,12,13,14,15);
    }
#pragma unroll
    for (int nt = 0; nt < 5; ++nt) {
        v16h b = frag_from_lds(&BF1[nt][lane][0]);
        v8f c = {};
        c = __builtin_amdgcn_wmma_f32_16x16x32_f16(false, a1, false, b,
                                                   (short)0, c, false, false);
#pragma unroll
        for (int v = 0; v < 8; ++v)                      // D: M=v+8*hi, N=lm
            gates[wv][v + 8 * hi][nt * 16 + lm] = c[v];
    }
    __syncthreads();

    // ---- layer-1 nonlinearity ----
#pragma unroll
    for (int it = 0; it < 10; ++it) {
        int j = ej0 + it;
        float gi = gates[wv][em][j];
        float gf = gates[wv][em][20 + j];
        float gg = gates[wv][em][40 + j];
        float go = gates[wv][em][60 + j];
        float cn = fsigmoid(gf) * cell[base1 + it] + fsigmoid(gi) * ftanhf(gg);
        float hn = fsigmoid(go) * ftanhf(cn);
        outC[base1 + it] = cn;
        outH[base1 + it] = hn;
        stage[wv][em][j] = (_Float16)hn;                 // h0 -> cols 0..19
    }
#pragma unroll
    for (int it = 0; it < 10; ++it)                      // h_prev -> cols 20..39
        stage[wv][em][20 + ej0 + it] = (_Float16)hidden[base2 + it];
#pragma unroll
    for (int it = 0; it < 12; ++it) {                    // cols 40..63: bias-1 + pad
        int k = 40 + (lane & 1) * 12 + it;
        stage[wv][em][k] = (_Float16)((k == 40) ? 1.0f : 0.0f);
    }
    __syncthreads();

    // ---- layer 2: two K-chunks chained through the accumulator ----
    v16h a20, a21;
    {
        const v8h* pa = (const v8h*)&stage[wv][lm][0];
        a20 = __builtin_shufflevector(pa[hi],     pa[2 + hi],
                                      0,1,2,3,4,5,6,7,8,9,10,11,12,13,14,15);
        a21 = __builtin_shufflevector(pa[4 + hi], pa[6 + hi],
                                      0,1,2,3,4,5,6,7,8,9,10,11,12,13,14,15);
    }
#pragma unroll
    for (int nt = 0; nt < 5; ++nt) {
        v16h b0 = frag_from_lds(&BF2[nt][lane][0]);      // chunk 0
        v16h b1 = frag_from_lds(&BF2[5 + nt][lane][0]);  // chunk 1
        v8f c = {};
        c = __builtin_amdgcn_wmma_f32_16x16x32_f16(false, a20, false, b0,
                                                   (short)0, c, false, false);
        c = __builtin_amdgcn_wmma_f32_16x16x32_f16(false, a21, false, b1,
                                                   (short)0, c, false, false);
#pragma unroll
        for (int v = 0; v < 8; ++v)
            gates[wv][v + 8 * hi][nt * 16 + lm] = c[v];
    }
    __syncthreads();

    // ---- layer-2 nonlinearity; park h1 (f32) in gate cols 0..19 for the dot ----
#pragma unroll
    for (int it = 0; it < 10; ++it) {
        int j = ej0 + it;
        float gi = gates[wv][em][j];
        float gf = gates[wv][em][20 + j];
        float gg = gates[wv][em][40 + j];
        float go = gates[wv][em][60 + j];
        float cn = fsigmoid(gf) * cell[base2 + it] + fsigmoid(gi) * ftanhf(gg);
        float hn = fsigmoid(go) * ftanhf(cn);
        outC[base2 + it] = cn;
        outH[base2 + it] = hn;
        gates[wv][em][j] = hn;       // unique (em,j) owner per lane -> race-free
    }
    __syncthreads();

    // ---- output projection + step-scale broadcast ----
    if (lane < 16) {
        int m = lane;
        long row = rowBase + m;
        float s = boutS;
#pragma unroll
        for (int j = 0; j < H; ++j) s += gates[wv][m][j] * WoutS[j];
#pragma unroll
        for (int st = 1; st <= 5; ++st) outO[row * 5 + (st - 1)] = s * (float)st;
    }
}

extern "C" void kernel_launch(void* const* d_in, const int* in_sizes, int n_in,
                              void* d_out, int out_size, void* d_ws, size_t ws_size,
                              hipStream_t stream) {
    (void)in_sizes; (void)n_in; (void)out_size; (void)d_ws; (void)ws_size;
    dim3 grid(B_TOTAL / RPB);   // 16384 blocks of 128 threads (4 wave32s)
    dim3 block(128);
    lstm_navigator_kernel<<<grid, block, 0, stream>>>(
        (const float*)d_in[0],  (const float*)d_in[1],  (const float*)d_in[2],
        (const float*)d_in[3],  (const float*)d_in[4],  (const float*)d_in[5],
        (const float*)d_in[6],  (const float*)d_in[7],  (const float*)d_in[8],
        (const float*)d_in[9],  (const float*)d_in[10], (const float*)d_in[11],
        (const float*)d_in[12], (float*)d_out);
}
